// MYGKT_88338887344573
// MI455X (gfx1250) — matrix operand order repaired and checked by reading, not compiled, not run on hardware
//
#include <hip/hip_runtime.h>
#include <hip/hip_bf16.h>
#include <math.h>
#include <stdint.h>

// Problem constants
#define Bq   32
#define Tq   100
#define NQ   128
#define NP   1000
#define Hd   64

typedef __attribute__((ext_vector_type(16))) __bf16 v16bf;
typedef __attribute__((ext_vector_type(8)))  float  v8f;
typedef __attribute__((ext_vector_type(4)))  unsigned int u32x4;
typedef __attribute__((ext_vector_type(8)))  int    i32x8;
typedef __attribute__((ext_vector_type(4)))  int    i32x4;

// ---------------- LDS layout (bytes) ----------------
#define OFF_HT    0        // float[128*64]    32768
#define OFF_M     32768    // float[128*64]    32768
#define OFF_A     65536    // bf16 [128*128]   32768  (A fragments, swizzled)
#define OFF_H1    98304    // bf16 [128*64]    16384  (layer-1 activations, swizzled)
#define OFF_GIB   114688   // bf16 [128*192]   49152  (GRU input gates)
#define OFF_W     163840   // bf16 [61440]    122880  (all weight B-fragments, staged once)
#define OFF_QOH   286720   // float[128]
#define OFF_SELF  287232   // float[128]
#define OFF_VOUT  287744   // float[64]
#define OFF_VIN   288000   // float[64]
#define OFF_AOUT  288256   // float[128]
#define OFF_AIN   288768   // float[128]
#define OFF_GH    289280   // float[192]
#define OFF_HPREV 290048   // float[64]
#define OFF_SS    290304   // float[128]
#define OFF_MISC  290816   // float[4]
#define LDS_BYTES 290944   // < 320KB per-WG CDNA5 limit; 1 WG / WGP

// Workspace layout: pemb f32[1000] @0 (4096B), then bf16 weight fragments (element offsets):
#define WSB_SELF1 0
#define WSB_OUT1H 8192
#define WSB_IN1H  16384
#define WSB_SELF2 24576
#define WSB_OUT2  28672
#define WSB_IN2   32768
#define WSB_IHT   36864
#define WSB_TOTAL 61440     // bf16 elements -> 122880 bytes -> 30720 dwords

__device__ __forceinline__ v8f wmma_bf16(v16bf a, v16bf b, v8f c) {
    return __builtin_amdgcn_wmma_f32_16x16x32_bf16(false, a, false, b, (short)0, c, false, false);
}

// ---------------- TDM: one-shot global->LDS weight staging ----------------
#if defined(__has_builtin)
#if __has_builtin(__builtin_amdgcn_tensor_load_to_lds) && __has_builtin(__builtin_amdgcn_s_wait_tensorcnt)
#define TDM_OK 1
#endif
#endif
#ifndef TDM_OK
#define TDM_OK 0
#endif

__device__ __forceinline__ void tdm_load_weights(const void* gsrc, unsigned lds_off, unsigned ndw) {
#if TDM_OK
    unsigned long long ga = (unsigned long long)(uintptr_t)gsrc;
    // D# group0: count=1 user desc, lds_addr, global_addr[56:0], type=2 ("image")
    u32x4 g0;
    g0[0] = 1u;
    g0[1] = lds_off;
    g0[2] = (unsigned)(ga & 0xFFFFFFFFu);
    g0[3] = (unsigned)((ga >> 32) & 0x01FFFFFFu) | (2u << 30);
    // D# group1: data_size=2 (4B elems); 2D tensor: dim0=ndw, dim1=1; tile ndw x 1
    i32x8 g1 = {};
    g1[0] = (int)(2u << 16);                        // wg_mask=0, data_size=4B
    g1[1] = (int)((ndw & 0xFFFFu) << 16);           // tensor_dim0 [63:48] low half
    g1[2] = (int)(((ndw >> 16) & 0xFFFFu))          // tensor_dim0 [79:64] high half
          | (int)(1u << 16);                        // tensor_dim1 [95:80] = 1 (low half)
    g1[3] = (int)((ndw & 0xFFFFu) << 16);           // tile_dim0 [127:112]
    g1[4] = 1;                                      // tile_dim1 [143:128] = 1
    g1[5] = (int)ndw;                               // tensor_dim0_stride [191:160]
    i32x4 z = {};
#if __clang_major__ >= 23
    i32x8 z8 = {};
    __builtin_amdgcn_tensor_load_to_lds(g0, g1, z, z, z8, 0);
#else
    __builtin_amdgcn_tensor_load_to_lds(g0, g1, z, z, 0);
#endif
    __builtin_amdgcn_s_wait_tensorcnt(0);
#else
    (void)gsrc; (void)lds_off; (void)ndw;
#endif
}

// ---- first-layer GEMM: (128x128 bf16 A frags, LDS) x (128x64 B frags, LDS)
//      -> relu(acc + bias[n]) stored bf16 into H1 in A-fragment layout (K=64)
__device__ __forceinline__ void gemm_l1(const __bf16* __restrict__ Af,
                                        const __bf16* __restrict__ Bf,
                                        const float*  __restrict__ bias,
                                        __bf16* __restrict__ H1,
                                        int wave, int lane) {
    int mt = wave;                       // 8 waves -> 8 row tiles
    v16bf a[4];
#pragma unroll
    for (int kt = 0; kt < 4; ++kt)
        a[kt] = *(const v16bf*)(Af + ((mt*4 + kt)*512 + lane*16));
    for (int nt = 0; nt < 4; ++nt) {
        v8f acc = {};
#pragma unroll
        for (int kt = 0; kt < 4; ++kt) {
            v16bf bfr = *(const v16bf*)(Bf + ((kt*4 + nt)*512 + lane*16));
            acc = wmma_bf16(a[kt], bfr, acc);
        }
        int n  = nt*16 + (lane & 15);
        float bv = bias[n];
        int h8 = (lane >> 4) * 8;
        int kk = n & 31, kt2 = n >> 5;
        int hh = (kk >> 3) & 1;
        int e  = (kk & 7) + ((kk >> 4) & 1) * 8;
#pragma unroll
        for (int i = 0; i < 8; ++i) {
            int m = mt*16 + h8 + i;
            float v = acc[i] + bv;
            v = v > 0.f ? v : 0.f;
            int ln2 = (m & 15) + hh*16;
            H1[((m >> 4)*2 + kt2)*512 + ln2*16 + e] = (__bf16)v;
        }
    }
}

// ---- second-layer GEMM: (128x64 H1 frags, LDS) x (64x64 B frags, LDS) -> s_M
// mode 0: M = aout[m]*v ; mode 1: M += ain[m]*v ; mode 2: if qoh[m] then M = v
__device__ __forceinline__ void gemm_l2(const __bf16* __restrict__ H1,
                                        const __bf16* __restrict__ Bf,
                                        const float*  __restrict__ bias,
                                        float* __restrict__ M,
                                        const float* __restrict__ rowscale,
                                        const float* __restrict__ qoh,
                                        int mode, int wave, int lane) {
    int mt = wave;
    v16bf a[2];
#pragma unroll
    for (int kt = 0; kt < 2; ++kt)
        a[kt] = *(const v16bf*)(H1 + ((mt*2 + kt)*512 + lane*16));
    for (int nt = 0; nt < 4; ++nt) {
        v8f acc = {};
#pragma unroll
        for (int kt = 0; kt < 2; ++kt) {
            v16bf bfr = *(const v16bf*)(Bf + ((kt*4 + nt)*512 + lane*16));
            acc = wmma_bf16(a[kt], bfr, acc);
        }
        int n  = nt*16 + (lane & 15);
        float bv = bias[n];
        int h8 = (lane >> 4) * 8;
#pragma unroll
        for (int i = 0; i < 8; ++i) {
            int m = mt*16 + h8 + i;
            float v = acc[i] + bv;
            if (mode == 0)      M[m*64 + n]  = rowscale[m]*v;
            else if (mode == 1) M[m*64 + n] += rowscale[m]*v;
            else if (qoh[m] > 0.5f) M[m*64 + n] = v;
        }
    }
}

// ---- GRU input GEMM: (128x128 A frags) x (128x192 W_ih^T frags, LDS) -> GI bf16 (+b_ih)
__device__ __forceinline__ void gemm_gi(const __bf16* __restrict__ Af,
                                        const __bf16* __restrict__ Bf,
                                        const float*  __restrict__ bih,
                                        __bf16* __restrict__ GIb,
                                        int wave, int lane) {
    int mt = wave;
    v16bf a[4];
#pragma unroll
    for (int kt = 0; kt < 4; ++kt)
        a[kt] = *(const v16bf*)(Af + ((mt*4 + kt)*512 + lane*16));
    for (int nt = 0; nt < 12; ++nt) {
        v8f acc = {};
#pragma unroll
        for (int kt = 0; kt < 4; ++kt) {
            v16bf bfr = *(const v16bf*)(Bf + ((kt*12 + nt)*512 + lane*16));
            acc = wmma_bf16(a[kt], bfr, acc);
        }
        int n  = nt*16 + (lane & 15);
        float bv = bih[n];
        int h8 = (lane >> 4) * 8;
#pragma unroll
        for (int i = 0; i < 8; ++i) {
            int m = mt*16 + h8 + i;
            GIb[m*192 + n] = (__bf16)(acc[i] + bv);
        }
    }
}

// =================== prep: weight fragment swizzle (f32 -> bf16 B-frags) ===================
__global__ void gkt_prep_w(const float* __restrict__ W_self1, const float* __restrict__ W_out1,
                           const float* __restrict__ W_in1,   const float* __restrict__ W_self2,
                           const float* __restrict__ W_out2,  const float* __restrict__ W_in2,
                           const float* __restrict__ W_ih,    __bf16* __restrict__ wsb) {
    int idx = blockIdx.x * blockDim.x + threadIdx.x;
    if (idx >= WSB_TOTAL) return;
    int seg, local, N;
    if      (idx < 8192)  { seg = 0; local = idx;          N = 64;  }
    else if (idx < 16384) { seg = 1; local = idx - 8192;   N = 64;  }
    else if (idx < 24576) { seg = 2; local = idx - 16384;  N = 64;  }
    else if (idx < 28672) { seg = 3; local = idx - 24576;  N = 64;  }
    else if (idx < 32768) { seg = 4; local = idx - 28672;  N = 64;  }
    else if (idx < 36864) { seg = 5; local = idx - 32768;  N = 64;  }
    else                  { seg = 6; local = idx - 36864;  N = 192; }
    int f = local >> 9, w = local & 511, ln = w >> 4, e = w & 15;
    int ntiles = N / 16;
    int kt = f / ntiles, nt = f % ntiles;
    int k = kt*32 + (e & 7) + ((ln >> 4) & 1)*8 + ((e >> 3) & 1)*16;   // B frag: table order, lanes 0-15 K=0..7,16..23, lanes 16-31 K=8..15,24..31
    int n = nt*16 + (ln & 15);
    float v;
    switch (seg) {
        case 0: v = W_self1[k*64 + n];        break;
        case 1: v = W_out1[(128 + k)*64 + n]; break;   // rows 2H..4H of W_out1
        case 2: v = W_in1[(128 + k)*64 + n];  break;
        case 3: v = W_self2[k*64 + n];        break;
        case 4: v = W_out2[k*64 + n];         break;
        case 5: v = W_in2[k*64 + n];          break;
        default: v = W_ih[n*128 + k];         break;   // B = W_ih^T
    }
    wsb[idx] = (__bf16)v;
}

// =================== prep: pemb_logit[p] = bias_p + pq_rel @ (q_emb_tab @ Wy2) ===================
__global__ void gkt_prep_pemb(const float* __restrict__ q_emb_tab, const float* __restrict__ W_outp,
                              const float* __restrict__ pq_rel,    const float* __restrict__ bias_p,
                              float* __restrict__ pemb) {
    __shared__ float u[NQ];
    int tid = threadIdx.x;
    if (tid < NQ) {
        float s = 0.f;
        for (int h = 0; h < Hd; ++h) s += q_emb_tab[tid*Hd + h] * W_outp[Hd + h];
        u[tid] = s;
    }
    __syncthreads();
    for (int pp = tid; pp < NP; pp += blockDim.x) {
        float s = bias_p[pp];
        const float* prow = pq_rel + (size_t)pp * NQ;
        for (int q = 0; q < NQ; ++q) s += prow[q] * u[q];
        pemb[pp] = s;
    }
}

// =================== main: one workgroup per batch element, whole T loop ===================
__global__ __launch_bounds__(256) void gkt_main(
    const int* __restrict__ p, const int* __restrict__ r,
    const float* __restrict__ adj, const float* __restrict__ pq_rel,
    const float* __restrict__ xq_emb, const float* __restrict__ q_emb_tab,
    const float* __restrict__ init_h,
    const float* __restrict__ b_self1, const float* __restrict__ b_self2,
    const float* __restrict__ b_out1,  const float* __restrict__ b_out2,
    const float* __restrict__ b_in1,   const float* __restrict__ b_in2,
    const float* __restrict__ b_ih,    const float* __restrict__ b_hh,
    const float* __restrict__ W_out1,  const float* __restrict__ W_in1,
    const float* __restrict__ W_hh,    const float* __restrict__ W_outp,
    const __bf16* __restrict__ wsb,    const float* __restrict__ pemb,
    float* __restrict__ y_out, float* __restrict__ h_out) {
    extern __shared__ char smem[];
    float*  s_ht   = (float*)(smem + OFF_HT);
    float*  s_M    = (float*)(smem + OFF_M);
    __bf16* s_A    = (__bf16*)(smem + OFF_A);
    __bf16* s_H1   = (__bf16*)(smem + OFF_H1);
    __bf16* s_GIb  = (__bf16*)(smem + OFF_GIB);
    __bf16* s_W    = (__bf16*)(smem + OFF_W);
    float*  s_qoh  = (float*)(smem + OFF_QOH);
    float*  s_self = (float*)(smem + OFF_SELF);
    float*  s_vout = (float*)(smem + OFF_VOUT);
    float*  s_vin  = (float*)(smem + OFF_VIN);
    float*  s_aout = (float*)(smem + OFF_AOUT);
    float*  s_ain  = (float*)(smem + OFF_AIN);
    float*  s_gh   = (float*)(smem + OFF_GH);
    float*  s_hp   = (float*)(smem + OFF_HPREV);
    float*  s_s    = (float*)(smem + OFF_SS);
    float*  s_misc = (float*)(smem + OFF_MISC);

    const __bf16* WBself1 = s_W + WSB_SELF1;
    const __bf16* WBout1h = s_W + WSB_OUT1H;
    const __bf16* WBin1h  = s_W + WSB_IN1H;
    const __bf16* WBself2 = s_W + WSB_SELF2;
    const __bf16* WBout2  = s_W + WSB_OUT2;
    const __bf16* WBin2   = s_W + WSB_IN2;
    const __bf16* WBihT   = s_W + WSB_IHT;

    const int b    = blockIdx.x;
    const int tid  = threadIdx.x;
    const int wave = tid >> 5;
    const int lane = tid & 31;

    // ---- stage all weight fragments into LDS once (TDM if available) ----
#if TDM_OK
    if (tid < 32) tdm_load_weights(wsb, OFF_W, WSB_TOTAL / 2);
#else
    {
        const unsigned* src = (const unsigned*)wsb;
        unsigned* dst = (unsigned*)(smem + OFF_W);
        for (int i = tid; i < WSB_TOTAL / 2; i += 256) dst[i] = src[i];
    }
#endif

    // init hidden state + h[b][0]
    for (int i = tid; i < NQ*Hd; i += 256) {
        float v = init_h[i];
        s_ht[i] = v;
        h_out[(size_t)b*(Tq+1)*NQ*Hd + i] = v;
    }
    __syncthreads();

    for (int t = 0; t < Tq; ++t) {
        const int pbt  = p[b*Tq + t];
        const int rsel = r[b*Tq + t];
        const float* xq = xq_emb + (size_t)rsel * NQ * Hd;  // xq_nc or xq_c
        if (tid < NQ) s_qoh[tid] = pq_rel[(size_t)pbt*NQ + tid];
        __syncthreads();
        if (tid == 0) {
            float s = 0.f;
            for (int q = 0; q < NQ; ++q) s += s_qoh[q];
            s_misc[0] = 1.f / (s + 1e-6f);
        }
        // build A fragments of ht_ = [ht | second]  (bf16, WMMA A layout)
        for (int idx = tid; idx < NQ*128; idx += 256) {
            int f = idx >> 9, w = idx & 511, ln = w >> 4, e = w & 15;
            int mt = f >> 2, kt = f & 3;
            int m  = mt*16 + (ln & 15);
            int kk = (e & 7) + ((ln >> 4) & 1)*8 + ((e >> 3) & 1)*16;
            int k  = kt*32 + kk;
            float v;
            if (k < Hd) v = s_ht[m*Hd + k];
            else {
                int j = k - Hd;
                v = (s_qoh[m] > 0.5f) ? xq[m*Hd + j] : q_emb_tab[m*Hd + j];
            }
            s_A[idx] = (__bf16)v;
        }
        __syncthreads();
        // self_ht (2H) and a_out/a_in
        if (tid < 128) {
            int d = tid; float s = 0.f;
            if (d < Hd) { for (int q = 0; q < NQ; ++q) s += s_qoh[q] * s_ht[q*Hd + d]; }
            else        { int j = d - Hd; for (int q = 0; q < NQ; ++q) s += s_qoh[q] * xq[q*Hd + j]; }
            s_self[d] = s * s_misc[0];
        } else {
            int q = tid - 128;
            float ao = 0.f, ai = 0.f;
            for (int qq = 0; qq < NQ; ++qq) {
                float wq = s_qoh[qq];
                if (wq != 0.f) { ao += wq * adj[qq*NQ + q]; ai += wq * adj[q*NQ + qq]; }
            }
            s_aout[q] = ao; s_ain[q] = ai;
        }
        __syncthreads();
        // fold self_ht through first 2H rows of W_out1 / W_in1 into bias vectors
        if (tid < Hd) {
            int j = tid; float v = b_out1[j];
            for (int d = 0; d < 128; ++d) v += s_self[d] * W_out1[d*Hd + j];
            s_vout[j] = v;
        } else if (tid < 128) {
            int j = tid - Hd; float v = b_in1[j];
            for (int d = 0; d < 128; ++d) v += s_self[d] * W_in1[d*Hd + j];
            s_vin[j] = v;
        }
        __syncthreads();
        // --- three MLPs via WMMA ---
        gemm_l1(s_A, WBout1h, s_vout,  s_H1, wave, lane); __syncthreads();
        gemm_l2(s_H1, WBout2, b_out2, s_M, s_aout, s_qoh, 0, wave, lane); __syncthreads();
        gemm_l1(s_A, WBin1h, s_vin,   s_H1, wave, lane); __syncthreads();
        gemm_l2(s_H1, WBin2,  b_in2,  s_M, s_ain,  s_qoh, 1, wave, lane); __syncthreads();
        gemm_l1(s_A, WBself1, b_self1, s_H1, wave, lane); __syncthreads();
        gemm_l2(s_H1, WBself2, b_self2, s_M, (const float*)0, s_qoh, 2, wave, lane); __syncthreads();
        // x_gru = [ht | m]: kt 0,1 of s_A are already ht; rewrite kt 2,3 with m
        for (int idx = tid; idx < NQ*128; idx += 256) {
            int f = idx >> 9, kt = f & 3;
            if (kt < 2) continue;
            int w = idx & 511, ln = w >> 4, e = w & 15;
            int mt = f >> 2;
            int m  = mt*16 + (ln & 15);
            int kk = (e & 7) + ((ln >> 4) & 1)*8 + ((e >> 3) & 1)*16;
            int j  = kt*32 + kk - Hd;
            s_A[idx] = (__bf16)s_M[m*Hd + j];
        }
        __syncthreads();
        gemm_gi(s_A, WBihT, b_ih, s_GIb, wave, lane);
        if (tid < Hd) s_hp[tid] = 0.f;
        __syncthreads();
        // sequential GRU over concepts (state = 64)
        for (int q = 0; q < NQ; ++q) {
            if (tid < 192) {
                float g = b_hh[tid];
                const float* wrow = W_hh + tid*Hd;
                for (int k2 = 0; k2 < Hd; ++k2) g += s_hp[k2] * wrow[k2];
                s_gh[tid] = g;
            }
            __syncthreads();
            if (tid < Hd) {
                int j = tid;
                const __bf16* gi = s_GIb + q*192;
                float rg = 1.f / (1.f + __expf(-((float)gi[j]      + s_gh[j])));
                float zg = 1.f / (1.f + __expf(-((float)gi[Hd + j] + s_gh[Hd + j])));
                float ng = tanhf((float)gi[128 + j] + rg * s_gh[128 + j]);
                float hn = (1.f - zg)*ng + zg*s_hp[j];
                s_hp[j] = hn;
                s_ht[q*Hd + j] = hn;
            }
            __syncthreads();
        }
        // s[q] = ht_new[q] . Wy1
        if (tid < NQ) {
            float s = 0.f;
            const float* hq = s_ht + tid*Hd;
            for (int j = 0; j < Hd; ++j) s += hq[j] * W_outp[j];
            s_s[tid] = s;
        }
        __syncthreads();
        // y[b,t,:] = sigmoid(pq_rel @ s + pemb)
        for (int pp = tid; pp < NP; pp += 256) {
            const float* prow = pq_rel + (size_t)pp * NQ;
            float lg = pemb[pp];
            for (int q = 0; q < NQ; ++q) lg += prow[q] * s_s[q];
            y_out[((size_t)b*Tq + t)*NP + pp] = 1.f / (1.f + __expf(-lg));
        }
        // h[b, t+1]
        for (int i = tid; i < NQ*Hd; i += 256)
            h_out[((size_t)b*(Tq+1) + (t+1))*NQ*Hd + i] = s_ht[i];
        __syncthreads();
    }
}

extern "C" void kernel_launch(void* const* d_in, const int* in_sizes, int n_in,
                              void* d_out, int out_size, void* d_ws, size_t ws_size,
                              hipStream_t stream) {
    (void)in_sizes; (void)n_in; (void)out_size; (void)ws_size;
    const int*   p        = (const int*)  d_in[0];
    const int*   r        = (const int*)  d_in[1];
    const float* adj      = (const float*)d_in[2];
    const float* pq_rel   = (const float*)d_in[3];
    const float* xq_emb   = (const float*)d_in[4];
    const float* q_emb    = (const float*)d_in[5];
    const float* init_h   = (const float*)d_in[6];
    const float* W_self1  = (const float*)d_in[7];
    const float* b_self1  = (const float*)d_in[8];
    const float* W_self2  = (const float*)d_in[9];
    const float* b_self2  = (const float*)d_in[10];
    const float* W_out1   = (const float*)d_in[11];
    const float* b_out1   = (const float*)d_in[12];
    const float* W_out2   = (const float*)d_in[13];
    const float* b_out2   = (const float*)d_in[14];
    const float* W_in1    = (const float*)d_in[15];
    const float* b_in1    = (const float*)d_in[16];
    const float* W_in2    = (const float*)d_in[17];
    const float* b_in2    = (const float*)d_in[18];
    const float* W_ih     = (const float*)d_in[19];
    const float* b_ih     = (const float*)d_in[20];
    const float* W_hh     = (const float*)d_in[21];
    const float* b_hh     = (const float*)d_in[22];
    const float* W_outp   = (const float*)d_in[23];
    const float* bias_p   = (const float*)d_in[24];

    float*  pemb = (float*)d_ws;                              // 1000 f32
    __bf16* wsb  = (__bf16*)((char*)d_ws + 4096);             // 61440 bf16 (~120KB); ws needs ~128KB

    gkt_prep_w<<<(WSB_TOTAL + 255) / 256, 256, 0, stream>>>(
        W_self1, W_out1, W_in1, W_self2, W_out2, W_in2, W_ih, wsb);
    gkt_prep_pemb<<<1, 256, 0, stream>>>(q_emb, W_outp, pq_rel, bias_p, pemb);

    (void)hipFuncSetAttribute(reinterpret_cast<const void*>(gkt_main),
                              hipFuncAttributeMaxDynamicSharedMemorySize, LDS_BYTES);

    float* y_out = (float*)d_out;                  // (32,100,1000)
    float* h_out = y_out + (size_t)Bq*Tq*NP;       // (32,101,128,64)
    gkt_main<<<Bq, 256, LDS_BYTES, stream>>>(
        p, r, adj, pq_rel, xq_emb, q_emb, init_h,
        b_self1, b_self2, b_out1, b_out2, b_in1, b_in2, b_ih, b_hh,
        W_out1, W_in1, W_hh, W_outp,
        wsb, pemb, y_out, h_out);
}
